// PointNet2SsgClassPcModel_44590350467469
// MI455X (gfx1250) — matrix operand order repaired and checked by reading, not compiled, not run on hardware
//
#include <hip/hip_runtime.h>

// PointNet++ SSG inference for MI455X (gfx1250, wave32, WMMA).
// All GEMMs use v_wmma_f32_16x16x32_f16 (f16 A/B, f32 accumulate).
// SA1/SA2 stages fully fused in LDS; SA3 + head are global WMMA GEMMs with
// 4x2 register-blocked tiles (8 WMMAs per K-step per wave).

#define EPSF 1e-5f

typedef __attribute__((ext_vector_type(16))) _Float16 v16h;
typedef __attribute__((ext_vector_type(8)))  float    v8f;

__device__ __forceinline__ v8f wmma_f16(v16h a, v16h b, v8f c) {
  // (neg_a, A, neg_b, B, c_mod, C, reuse_a, reuse_b)
  return __builtin_amdgcn_wmma_f32_16x16x32_f16(false, a, false, b, (short)0, c,
                                                false, false);
}

__device__ __forceinline__ unsigned ballot32(bool p) {
#if __has_builtin(__builtin_amdgcn_ballot_w32)
  return __builtin_amdgcn_ballot_w32(p);
#else
  return (unsigned)__ballot(p);
#endif
}

// A fragment, 16-bit A 16x32 layout: lanes 0-15 rows M=0..15 hold K kb+{0..7},
// kb+16+{0..7} with kb=0; lanes 16-31 same rows with kb=8.
__device__ __forceinline__ v16h load_a_frag(const _Float16* A, int lda,
                                            int rowBase, int k0, int lane) {
  const int row = rowBase + (lane & 15);
  const int kb  = (lane >> 4) * 8;
  const _Float16* p = A + (size_t)row * lda + k0 + kb;
  v16h a;
#pragma unroll
  for (int j = 0; j < 8; ++j) { a[j] = p[j]; a[8 + j] = p[16 + j]; }
  return a;
}

// B fragment, 32x16 B: lane holds column N=lane&15, K = kb+{0..7}, kb+16+{0..7},
// kb = (lane>=16)*8. Weights stored transposed [N][Kpad] -> contiguous.
__device__ __forceinline__ v16h load_b_frag(const _Float16* Wt, int kpad,
                                            int colBase, int k0, int lane) {
  const int col = colBase + (lane & 15);
  const int kb  = (lane >> 4) * 8;
  const _Float16* p = Wt + (size_t)col * kpad + k0 + kb;
  v16h b;
#pragma unroll
  for (int j = 0; j < 8; ++j) { b[j] = p[j]; b[8 + j] = p[16 + j]; }
  return b;
}

// LDS A [M][K] f16  x  Wt [N][K] f16  -> LDS Out [M][N] f16 (bias + relu).
__device__ __forceinline__ void gemm_lds(const _Float16* A, int M, int K,
                                         const _Float16* __restrict__ Wt,
                                         const float* __restrict__ bias,
                                         _Float16* Out, int N) {
  const int lane = threadIdx.x & 31;
  const int wave = threadIdx.x >> 5;
  const int nw   = blockDim.x >> 5;
  const int nt   = N >> 4;
  const int tiles = (M >> 4) * nt;
  for (int t = wave; t < tiles; t += nw) {
    const int tm = t / nt, tn = t - tm * nt;
    v8f c = {};
    for (int k0 = 0; k0 < K; k0 += 32) {
      v16h a = load_a_frag(A, K, tm * 16, k0, lane);
      v16h b = load_b_frag(Wt, K, tn * 16, k0, lane);
      c = wmma_f16(a, b, c);
    }
    const int col = tn * 16 + (lane & 15);
    const int rb  = tm * 16 + ((lane >> 4) << 3);  // C rows: lanes>=16 -> +8
    const float bs = bias[col];
#pragma unroll
    for (int r = 0; r < 8; ++r) {
      float v = c[r] + bs;
      v = v > 0.f ? v : 0.f;
      Out[(rb + r) * N + col] = (_Float16)v;
    }
  }
}

// Last layer of a stage: fuse relu + row-max into pmax[(M/8)][N] partials.
__device__ __forceinline__ void gemm_lds_max(const _Float16* A, int M, int K,
                                             const _Float16* __restrict__ Wt,
                                             const float* __restrict__ bias,
                                             float* pmax, int N) {
  const int lane = threadIdx.x & 31;
  const int wave = threadIdx.x >> 5;
  const int nw   = blockDim.x >> 5;
  const int nt   = N >> 4;
  const int tiles = (M >> 4) * nt;
  for (int t = wave; t < tiles; t += nw) {
    const int tm = t / nt, tn = t - tm * nt;
    v8f c = {};
    for (int k0 = 0; k0 < K; k0 += 32) {
      v16h a = load_a_frag(A, K, tm * 16, k0, lane);
      v16h b = load_b_frag(Wt, K, tn * 16, k0, lane);
      c = wmma_f16(a, b, c);
    }
    const int col = tn * 16 + (lane & 15);
    const float bs = bias[col];
    float m = 0.f;  // relu floor
#pragma unroll
    for (int r = 0; r < 8; ++r) {
      float v = c[r] + bs;
      m = v > m ? v : m;
    }
    pmax[(tm * 2 + (lane >> 4)) * N + col] = m;  // unique slot per (tile,half)
  }
}

// Global GEMM, 4x2 register-blocked: each wave owns a 64x32 macro-tile
// (8 accumulators, 8 WMMAs per K-step, A frags reused x2, B frags x4).
// Requires M % 64 == 0, N % 32 == 0, K % 32 == 0.
__global__ __launch_bounds__(256) void gemm_global(
    const _Float16* __restrict__ A, int K, const _Float16* __restrict__ Wt,
    const float* __restrict__ bias, void* __restrict__ outp, int M, int N,
    int Nvalid, int ldo, int relu, int f32out) {
  const int lane  = threadIdx.x & 31;
  const int wid   = blockIdx.x * (blockDim.x >> 5) + (threadIdx.x >> 5);
  const int nwave = gridDim.x * (blockDim.x >> 5);
  const int ntn   = N >> 5;                 // macro cols (32 wide)
  const int tiles = (M >> 6) * ntn;         // macro rows (64 tall)
  for (int t = wid; t < tiles; t += nwave) {
    const int tm = t / ntn, tn = t - tm * ntn;
    const int mb = tm * 64, nb = tn * 32;
    v8f c[4][2];
#pragma unroll
    for (int i = 0; i < 4; ++i)
#pragma unroll
      for (int j = 0; j < 2; ++j) c[i][j] = (v8f){};
    for (int k0 = 0; k0 < K; k0 += 32) {
      if (k0 + 32 < K)  // prefetch next K-slab of the A panel (L2-resident)
        __builtin_prefetch(A + (size_t)(mb + (lane & 15)) * K + k0 + 32, 0, 1);
      v16h a0 = load_a_frag(A, K, mb +  0, k0, lane);
      v16h a1 = load_a_frag(A, K, mb + 16, k0, lane);
      v16h a2 = load_a_frag(A, K, mb + 32, k0, lane);
      v16h a3 = load_a_frag(A, K, mb + 48, k0, lane);
      v16h b0 = load_b_frag(Wt, K, nb +  0, k0, lane);
      v16h b1 = load_b_frag(Wt, K, nb + 16, k0, lane);
      c[0][0] = wmma_f16(a0, b0, c[0][0]);
      c[0][1] = wmma_f16(a0, b1, c[0][1]);
      c[1][0] = wmma_f16(a1, b0, c[1][0]);
      c[1][1] = wmma_f16(a1, b1, c[1][1]);
      c[2][0] = wmma_f16(a2, b0, c[2][0]);
      c[2][1] = wmma_f16(a2, b1, c[2][1]);
      c[3][0] = wmma_f16(a3, b0, c[3][0]);
      c[3][1] = wmma_f16(a3, b1, c[3][1]);
    }
#pragma unroll
    for (int i = 0; i < 4; ++i) {
      const int rb = mb + i * 16 + ((lane >> 4) << 3);
#pragma unroll
      for (int j = 0; j < 2; ++j) {
        const int col = nb + j * 16 + (lane & 15);
        if (col < Nvalid) {
          const float bs = bias[col];
#pragma unroll
          for (int r = 0; r < 8; ++r) {
            float v = c[i][j][r] + bs;
            if (relu) v = v > 0.f ? v : 0.f;
            if (f32out)
              ((float*)outp)[(size_t)(rb + r) * ldo + col] = v;
            else
              ((_Float16*)outp)[(size_t)(rb + r) * ldo + col] = (_Float16)v;
          }
        }
      }
    }
  }
}

// Fold BN into weights, transpose to [Npad][Kpad] f16, emit f32 bias[Npad].
__global__ void fold_kernel(const float* __restrict__ w, const float* __restrict__ b,
                            const float* __restrict__ gamma, const float* __restrict__ beta,
                            const float* __restrict__ mean, const float* __restrict__ var,
                            _Float16* __restrict__ wt, float* __restrict__ bias,
                            int K, int N, int Kpad, int Npad) {
  const int total = Kpad * Npad;
  for (int i = blockIdx.x * blockDim.x + threadIdx.x; i < total;
       i += gridDim.x * blockDim.x) {
    const int n = i / Kpad, k = i - n * Kpad;
    float v = 0.f;
    if (k < K && n < N) {
      const float s = gamma ? gamma[n] * rsqrtf(var[n] + EPSF) : 1.f;
      v = w[(size_t)k * N + n] * s;
    }
    wt[i] = (_Float16)v;
    if (k == 0) {
      float bb = 0.f;
      if (n < N) {
        if (gamma) {
          const float s = gamma[n] * rsqrtf(var[n] + EPSF);
          bb = (b[n] - mean[n]) * s + beta[n];
        } else {
          bb = b[n];
        }
      }
      bias[n] = bb;
    }
  }
}

// SA1 layer-1 (K=3) stays scalar: fold to f32 [3][64] + bias[64].
__global__ void fold1_kernel(const float* __restrict__ w, const float* __restrict__ b,
                             const float* __restrict__ gamma, const float* __restrict__ beta,
                             const float* __restrict__ mean, const float* __restrict__ var,
                             float* __restrict__ wf, float* __restrict__ bias) {
  const int i = blockIdx.x * blockDim.x + threadIdx.x;
  if (i < 192) {
    const int k = i / 64, n = i - k * 64;
    const float s = gamma[n] * rsqrtf(var[n] + EPSF);
    wf[i] = w[i] * s;  // w is [3][64] row-major
    if (k == 0) bias[n] = (b[n] - mean[n]) * s + beta[n];
  }
}

// Farthest point sampling: one block per batch, sequential npoint iterations.
__global__ __launch_bounds__(256) void fps_kernel(const float* __restrict__ xyz,
                                                  int n, int npoint,
                                                  int* __restrict__ out) {
  __shared__ float dist[1024];
  __shared__ float rv[256];
  __shared__ int   ri[256];
  __shared__ int   sFar;
  const int b = blockIdx.x, t = threadIdx.x;
  const float* base = xyz + (size_t)b * n * 3;
  for (int i = t; i < n; i += 256) dist[i] = 1e10f;
  if (t == 0) sFar = 0;
  __syncthreads();
  for (int it = 0; it < npoint; ++it) {
    const int far = sFar;
    if (t == 0) out[(size_t)b * npoint + it] = far;
    const float cx = base[far * 3], cy = base[far * 3 + 1], cz = base[far * 3 + 2];
    float bestv = -1.f;
    int   besti = 0;
    for (int i = t; i < n; i += 256) {
      const float dx = base[i * 3] - cx, dy = base[i * 3 + 1] - cy,
                  dz = base[i * 3 + 2] - cz;
      const float d  = dx * dx + dy * dy + dz * dz;
      const float nd = fminf(dist[i], d);
      dist[i] = nd;
      if (nd > bestv) { bestv = nd; besti = i; }
    }
    rv[t] = bestv; ri[t] = besti;
    __syncthreads();
    for (int s = 128; s > 0; s >>= 1) {
      if (t < s) {
        if (rv[t + s] > rv[t] ||
            (rv[t + s] == rv[t] && ri[t + s] < ri[t])) {
          rv[t] = rv[t + s]; ri[t] = ri[t + s];
        }
      }
      __syncthreads();
    }
    if (t == 0) sFar = ri[0];
    __syncthreads();
  }
}

// Fused SA1: ball query (r=0.2, ns=32) + MLP 3->64->64->128 + maxpool.
__global__ __launch_bounds__(256) void sa1_kernel(
    const float* __restrict__ xyz, const int* __restrict__ fps1,
    const float* __restrict__ w1, const float* __restrict__ b1,
    const _Float16* __restrict__ wt2, const float* __restrict__ bias2,
    const _Float16* __restrict__ wt3, const float* __restrict__ bias3,
    _Float16* __restrict__ feats1, float* __restrict__ newxyz) {
  __shared__ float cen[3];
  __shared__ int   idxbuf[32];
  __shared__ int   scnt;
  __shared__ float relx[32][3];
  __shared__ float w1s[192], b1s[64];
  __shared__ _Float16 actA[32 * 64];
  __shared__ _Float16 actB[32 * 64];
  __shared__ float pmax[4 * 128];
  const int t = threadIdx.x, g = blockIdx.x;
  const int b = g >> 9;
  const float* pb = xyz + (size_t)b * 1024 * 3;
  if (t == 0) {
    const int f = fps1[g];
    cen[0] = pb[f * 3]; cen[1] = pb[f * 3 + 1]; cen[2] = pb[f * 3 + 2];
    newxyz[(size_t)g * 3 + 0] = cen[0];
    newxyz[(size_t)g * 3 + 1] = cen[1];
    newxyz[(size_t)g * 3 + 2] = cen[2];
  }
  for (int i = t; i < 192; i += 256) w1s[i] = w1[i];
  if (t < 64) b1s[t] = b1[t];
  __syncthreads();
  if (t < 32) {  // wave0: index-ordered ball-query compaction via ballot
    int found = 0;
    const float cx = cen[0], cy = cen[1], cz = cen[2];
    for (int base = 0; base < 1024 && found < 32; base += 32) {
      const int p = base + t;
      const float dx = pb[p * 3] - cx, dy = pb[p * 3 + 1] - cy,
                  dz = pb[p * 3 + 2] - cz;
      const bool in = (dx * dx + dy * dy + dz * dz) <= 0.04f;
      const unsigned m = ballot32(in);
      const int pos = found + __popc(m & ((1u << t) - 1u));
      if (in && pos < 32) idxbuf[pos] = p;
      found += __popc(m);
    }
    if (t == 0) scnt = found < 32 ? found : 32;
  }
  __syncthreads();
  int pown = 0;
  if (t < 32) pown = idxbuf[(t < scnt) ? t : 0];  // pad with first
  __syncthreads();
  if (t < 32) {
    idxbuf[t] = pown;
    relx[t][0] = pb[pown * 3] - cen[0];
    relx[t][1] = pb[pown * 3 + 1] - cen[1];
    relx[t][2] = pb[pown * 3 + 2] - cen[2];
  }
  __syncthreads();
  for (int i = t; i < 32 * 64; i += 256) {  // layer1, K=3 scalar
    const int row = i >> 6, col = i & 63;
    float v = b1s[col] + relx[row][0] * w1s[col] + relx[row][1] * w1s[64 + col] +
              relx[row][2] * w1s[128 + col];
    actA[i] = (_Float16)(v > 0.f ? v : 0.f);
  }
  __syncthreads();
  gemm_lds(actA, 32, 64, wt2, bias2, actB, 64);
  __syncthreads();
  gemm_lds_max(actB, 32, 64, wt3, bias3, pmax, 128);
  __syncthreads();
  for (int col = t; col < 128; col += 256) {
    float m = pmax[col];
    for (int j = 1; j < 4; ++j) m = fmaxf(m, pmax[j * 128 + col]);
    feats1[(size_t)g * 128 + col] = (_Float16)m;
  }
}

// Fused SA2: ball query (r=0.4, ns=64) + MLP 131(->160)->128->128->256 + maxpool.
__global__ __launch_bounds__(256) void sa2_kernel(
    const float* __restrict__ xyz1, const int* __restrict__ fps2,
    const _Float16* __restrict__ feats1,
    const _Float16* __restrict__ wt1, const float* __restrict__ bias1,
    const _Float16* __restrict__ wt2, const float* __restrict__ bias2,
    const _Float16* __restrict__ wt3, const float* __restrict__ bias3,
    _Float16* __restrict__ feats2, float* __restrict__ newxyz2) {
  __shared__ float cen[3];
  __shared__ int   idxbuf[64];
  __shared__ int   scnt;
  __shared__ _Float16 smem[64 * 160 + 64 * 128];  // A0 | A1 ; A2 reuses A0
  __shared__ float pmax[8 * 256];
  const int t = threadIdx.x, g = blockIdx.x;
  const int b = g >> 7;
  const float* pb = xyz1 + (size_t)b * 512 * 3;
  if (t == 0) {
    const int f = fps2[g];
    cen[0] = pb[f * 3]; cen[1] = pb[f * 3 + 1]; cen[2] = pb[f * 3 + 2];
    newxyz2[(size_t)g * 3 + 0] = cen[0];
    newxyz2[(size_t)g * 3 + 1] = cen[1];
    newxyz2[(size_t)g * 3 + 2] = cen[2];
  }
  __syncthreads();
  if (t < 32) {
    int found = 0;
    const float cx = cen[0], cy = cen[1], cz = cen[2];
    for (int base = 0; base < 512 && found < 64; base += 32) {
      const int p = base + t;
      const float dx = pb[p * 3] - cx, dy = pb[p * 3 + 1] - cy,
                  dz = pb[p * 3 + 2] - cz;
      const bool in = (dx * dx + dy * dy + dz * dz) <= 0.16f;
      const unsigned m = ballot32(in);
      const int pos = found + __popc(m & ((1u << t) - 1u));
      if (in && pos < 64) idxbuf[pos] = p;
      found += __popc(m);
    }
    if (t == 0) scnt = found < 64 ? found : 64;
  }
  __syncthreads();
  int pown = 0;
  if (t < 64) pown = idxbuf[(t < scnt) ? t : 0];
  __syncthreads();
  if (t < 64) idxbuf[t] = pown;
  __syncthreads();
  _Float16* A0 = smem;
  _Float16* A1 = smem + 64 * 160;
  _Float16* A2 = smem;  // A0 dead after layer 1
  for (int i = t; i < 64 * 160; i += 256) {  // gather rel-xyz + feats1, pad K
    const int row = i / 160, c = i - row * 160;
    const int p = idxbuf[row];
    float v;
    if (c < 3)        v = pb[p * 3 + c] - cen[c];
    else if (c < 131) v = (float)feats1[((size_t)b * 512 + p) * 128 + (c - 3)];
    else              v = 0.f;
    A0[i] = (_Float16)v;
  }
  __syncthreads();
  gemm_lds(A0, 64, 160, wt1, bias1, A1, 128);
  __syncthreads();
  gemm_lds(A1, 64, 128, wt2, bias2, A2, 128);
  __syncthreads();
  gemm_lds_max(A2, 64, 128, wt3, bias3, pmax, 256);
  __syncthreads();
  for (int col = t; col < 256; col += 256) {
    float m = pmax[col];
    for (int j = 1; j < 8; ++j) m = fmaxf(m, pmax[j * 256 + col]);
    feats2[(size_t)g * 256 + col] = (_Float16)m;
  }
}

// SA3 input build: [8192][288] = concat(xyz2(3), feats2(256), pad(29)).
__global__ void sa3_build(const float* __restrict__ newxyz2,
                          const _Float16* __restrict__ feats2,
                          _Float16* __restrict__ A0) {
  const int i = blockIdx.x * blockDim.x + threadIdx.x;
  if (i >= 8192 * 288) return;
  const int row = i / 288, c = i - row * 288;
  float v;
  if (c < 3)        v = newxyz2[(size_t)row * 3 + c];
  else if (c < 259) v = (float)feats2[(size_t)row * 256 + (c - 3)];
  else              v = 0.f;
  A0[i] = (_Float16)v;
}

// Global maxpool over 128 points per batch: [64*128][1024] -> [64][1024].
__global__ void maxpool3(const _Float16* __restrict__ in,
                         _Float16* __restrict__ out) {
  const int i = blockIdx.x * blockDim.x + threadIdx.x;
  if (i >= 64 * 1024) return;
  const int b = i >> 10, c = i & 1023;
  float m = 0.f;  // relu outputs are >= 0
  for (int p = 0; p < 128; ++p)
    m = fmaxf(m, (float)in[((size_t)(b * 128 + p)) * 1024 + c]);
  out[i] = (_Float16)m;
}

extern "C" void kernel_launch(void* const* d_in, const int* in_sizes, int n_in,
                              void* d_out, int out_size, void* d_ws, size_t ws_size,
                              hipStream_t stream) {
  (void)in_sizes; (void)n_in; (void)out_size; (void)ws_size;
  const float* pos = (const float*)d_in[0];
  // param layout: layer dicts flatten as w,b,gamma,beta,mean,var
  // sa1: 1,7,13  sa2: 19,25,31  sa3: 37,43,49  head: 55,61,67(w,b only)
  auto F = [&](int i) { return (const float*)d_in[i]; };

  char* ws = (char*)d_ws;
  size_t off = 0;
  auto alloc = [&](size_t bytes) -> char* {
    char* p = ws + off;
    off = (off + bytes + 255) & ~(size_t)255;
    return p;
  };
  int*       fps1   = (int*)alloc(64 * 512 * 4);
  int*       fps2   = (int*)alloc(64 * 128 * 4);
  float*     nx1    = (float*)alloc(64 * 512 * 3 * 4);
  float*     nx2    = (float*)alloc(64 * 128 * 3 * 4);
  _Float16*  feats1 = (_Float16*)alloc((size_t)64 * 512 * 128 * 2);
  _Float16*  feats2 = (_Float16*)alloc((size_t)64 * 128 * 256 * 2);
  _Float16*  a0_3   = (_Float16*)alloc((size_t)8192 * 288 * 2);
  _Float16*  b1_3   = (_Float16*)alloc((size_t)8192 * 256 * 2);
  _Float16*  b2_3   = (_Float16*)alloc((size_t)8192 * 1024 * 2);
  _Float16*  b3_3   = (_Float16*)alloc((size_t)8192 * 1024 * 2);
  _Float16*  gfeat  = (_Float16*)alloc(64 * 1024 * 2);
  _Float16*  h1     = (_Float16*)alloc(64 * 512 * 2);
  _Float16*  h2     = (_Float16*)alloc(64 * 256 * 2);
  float*     w1f    = (float*)alloc(192 * 4);
  float*     b1f    = (float*)alloc(64 * 4);
  _Float16*  wt12 = (_Float16*)alloc(64 * 64 * 2);    float* bb12 = (float*)alloc(64 * 4);
  _Float16*  wt13 = (_Float16*)alloc(128 * 64 * 2);   float* bb13 = (float*)alloc(128 * 4);
  _Float16*  wt21 = (_Float16*)alloc(128 * 160 * 2);  float* bb21 = (float*)alloc(128 * 4);
  _Float16*  wt22 = (_Float16*)alloc(128 * 128 * 2);  float* bb22 = (float*)alloc(128 * 4);
  _Float16*  wt23 = (_Float16*)alloc(256 * 128 * 2);  float* bb23 = (float*)alloc(256 * 4);
  _Float16*  wt31 = (_Float16*)alloc(256 * 288 * 2);  float* bb31 = (float*)alloc(256 * 4);
  _Float16*  wt32 = (_Float16*)alloc((size_t)1024 * 256 * 2);  float* bb32 = (float*)alloc(1024 * 4);
  _Float16*  wt33 = (_Float16*)alloc((size_t)1024 * 1024 * 2); float* bb33 = (float*)alloc(1024 * 4);
  _Float16*  wth1 = (_Float16*)alloc((size_t)512 * 1024 * 2);  float* bbh1 = (float*)alloc(512 * 4);
  _Float16*  wth2 = (_Float16*)alloc((size_t)256 * 512 * 2);   float* bbh2 = (float*)alloc(256 * 4);
  _Float16*  wth3 = (_Float16*)alloc(64 * 256 * 2);            float* bbh3 = (float*)alloc(64 * 4);

  // --- weight folding / transpose / pad ---
  auto fold = [&](int base, _Float16* wt, float* bias, int K, int N, int Kp,
                  int Np, bool bn) {
    const int total = Kp * Np;
    const int blocks = (total + 255) / 256;
    fold_kernel<<<blocks, 256, 0, stream>>>(
        F(base), F(base + 1), bn ? F(base + 2) : nullptr,
        bn ? F(base + 3) : nullptr, bn ? F(base + 4) : nullptr,
        bn ? F(base + 5) : nullptr, wt, bias, K, N, Kp, Np);
  };
  fold1_kernel<<<1, 256, 0, stream>>>(F(1), F(2), F(3), F(4), F(5), F(6), w1f, b1f);
  fold(7,  wt12, bb12, 64,   64,   64,   64,   true);
  fold(13, wt13, bb13, 64,   128,  64,   128,  true);
  fold(19, wt21, bb21, 131,  128,  160,  128,  true);
  fold(25, wt22, bb22, 128,  128,  128,  128,  true);
  fold(31, wt23, bb23, 128,  256,  128,  256,  true);
  fold(37, wt31, bb31, 259,  256,  288,  256,  true);
  fold(43, wt32, bb32, 256,  1024, 256,  1024, true);
  fold(49, wt33, bb33, 1024, 1024, 1024, 1024, true);
  fold(55, wth1, bbh1, 1024, 512,  1024, 512,  true);
  fold(61, wth2, bbh2, 512,  256,  512,  256,  true);
  fold(67, wth3, bbh3, 256,  40,   256,  64,   false);

  // --- SA1 ---
  fps_kernel<<<64, 256, 0, stream>>>(pos, 1024, 512, fps1);
  sa1_kernel<<<64 * 512, 256, 0, stream>>>(pos, fps1, w1f, b1f, wt12, bb12,
                                           wt13, bb13, feats1, nx1);
  // --- SA2 ---
  fps_kernel<<<64, 256, 0, stream>>>(nx1, 512, 128, fps2);
  sa2_kernel<<<64 * 128, 256, 0, stream>>>(nx1, fps2, feats1, wt21, bb21, wt22,
                                           bb22, wt23, bb23, feats2, nx2);
  // --- SA3 (global, 64x32 macro-tiles per wave) ---
  sa3_build<<<(8192 * 288 + 255) / 256, 256, 0, stream>>>(nx2, feats2, a0_3);
  {
    int tiles = (8192 / 64) * (256 / 32);
    gemm_global<<<(tiles + 7) / 8, 256, 0, stream>>>(a0_3, 288, wt31, bb31,
                                                     b1_3, 8192, 256, 256, 256, 1, 0);
    tiles = (8192 / 64) * (1024 / 32);
    gemm_global<<<(tiles + 7) / 8, 256, 0, stream>>>(b1_3, 256, wt32, bb32,
                                                     b2_3, 8192, 1024, 1024, 1024, 1, 0);
    gemm_global<<<(tiles + 7) / 8, 256, 0, stream>>>(b2_3, 1024, wt33, bb33,
                                                     b3_3, 8192, 1024, 1024, 1024, 1, 0);
  }
  maxpool3<<<(64 * 1024 + 255) / 256, 256, 0, stream>>>(b3_3, gfeat);
  // --- head (M=64 -> one macro row) ---
  {
    int tiles = (64 / 64) * (512 / 32);
    gemm_global<<<(tiles + 7) / 8, 256, 0, stream>>>(gfeat, 1024, wth1, bbh1,
                                                     h1, 64, 512, 512, 512, 1, 0);
    tiles = (64 / 64) * (256 / 32);
    gemm_global<<<(tiles + 7) / 8, 256, 0, stream>>>(h1, 512, wth2, bbh2, h2,
                                                     64, 256, 256, 256, 1, 0);
    tiles = (64 / 64) * (64 / 32);
    gemm_global<<<(tiles + 7) / 8, 256, 0, stream>>>(h2, 256, wth3, bbh3,
                                                     d_out, 64, 64, 40, 40, 0, 1);
  }
}
